// RandomFeatureAttention_67405216743939
// MI455X (gfx1250) — compile-verified, matrix operand order checked by reading, not run on hardware
//
#include <hip/hip_runtime.h>
#include <hip/hip_bf16.h>
#include <math.h>
#include <stdint.h>

// ---------------------------------------------------------------------------
// RandomFeatureAttention for MI455X (gfx1250), wave32 + WMMA f32_16x16x32_f16
//  - double-buffered LDS pipeline in the big GEMMs
//  - global_load_async_to_lds_b128 (ASYNCcnt) staging in the kv reduction
// ---------------------------------------------------------------------------

typedef __attribute__((ext_vector_type(16))) _Float16 v16h;
typedef __attribute__((ext_vector_type(8)))  _Float16 half8;
typedef __attribute__((ext_vector_type(8)))  float    v8f;

#define B_SZ   4
#define N_SEQ  4096
#define E_DIM  1024
#define H_NUM  16
#define D_DIM  64
#define P_DIM  64
#define F_DIM  128          // 2 * P_DIM
#define EPS    1e-4f

static __device__ inline v8f wmma_f16(v16h a, v16h b, v8f c) {
  return __builtin_amdgcn_wmma_f32_16x16x32_f16(
      /*neg_a=*/false, a, /*neg_b=*/false, b,
      /*c_mod=*/(short)0, c, /*reuse_a=*/false, /*reuse_b=*/false);
}

// ===========================================================================
// Kernel 1/5: C[M,N] = A[M,K] @ W[N,K]^T + bias[N]   (fp32 in, fp32 out)
// Block tile 128x64, K-step 32, 256 threads = 8 waves, each wave 32x32.
// Double-buffered LDS: global loads for tile k+1 overlap WMMAs of tile k.
// ===========================================================================
__global__ __launch_bounds__(256)
void gemm_xWt_bias(const float* __restrict__ A, const float* __restrict__ W,
                   const float* __restrict__ bias, float* __restrict__ C,
                   int M, int N, int K) {
  __shared__ _Float16 sA[2][128][40];   // row stride 80B (16B aligned)
  __shared__ _Float16 sB[2][64][40];

  const int tid  = threadIdx.x;
  const int lane = tid & 31;
  const int wave = tid >> 5;
  const int nBlk = N >> 6;                       // N / 64
  const int bm   = (blockIdx.x / nBlk) << 7;     // * 128
  const int bn   = (blockIdx.x % nBlk) << 6;     // * 64
  const int wm   = (wave & 3) << 5;              // 0,32,64,96
  const int wn   = (wave >> 2) << 5;             // 0,32

  v8f acc[2][2] = {};

  const int lhalf = lane & 15;
  const int kb0   = (lane >> 4) << 3;            // 0 or 8

  float4 ra[4], rb[2];

  auto loadAB = [&](int k0) {
    #pragma unroll
    for (int i = 0; i < 4; ++i) {
      int idx = i * 256 + tid;                   // 0..1023
      int r   = idx >> 3;
      int c4  = (idx & 7) << 2;
      ra[i] = *(const float4*)&A[(size_t)(bm + r) * K + k0 + c4];
    }
    #pragma unroll
    for (int i = 0; i < 2; ++i) {
      int idx = i * 256 + tid;                   // 0..511
      int r   = idx >> 3;
      int c4  = (idx & 7) << 2;
      rb[i] = *(const float4*)&W[(size_t)(bn + r) * K + k0 + c4];
    }
  };
  auto storeAB = [&](int buf) {
    #pragma unroll
    for (int i = 0; i < 4; ++i) {
      int idx = i * 256 + tid;
      int r   = idx >> 3;
      int c4  = (idx & 7) << 2;
      sA[buf][r][c4 + 0] = (_Float16)ra[i].x;
      sA[buf][r][c4 + 1] = (_Float16)ra[i].y;
      sA[buf][r][c4 + 2] = (_Float16)ra[i].z;
      sA[buf][r][c4 + 3] = (_Float16)ra[i].w;
    }
    #pragma unroll
    for (int i = 0; i < 2; ++i) {
      int idx = i * 256 + tid;
      int r   = idx >> 3;
      int c4  = (idx & 7) << 2;
      sB[buf][r][c4 + 0] = (_Float16)rb[i].x;
      sB[buf][r][c4 + 1] = (_Float16)rb[i].y;
      sB[buf][r][c4 + 2] = (_Float16)rb[i].z;
      sB[buf][r][c4 + 3] = (_Float16)rb[i].w;
    }
  };

  // prologue: stage tile 0
  loadAB(0);
  storeAB(0);
  __syncthreads();

  const int nK = K >> 5;
  for (int kt = 0; kt < nK; ++kt) {
    const int cur = kt & 1;
    const bool hasNext = (kt + 1) < nK;
    if (hasNext) loadAB((kt + 1) << 5);          // global loads overlap WMMAs

    v16h af[2], bf[2];
    #pragma unroll
    for (int t = 0; t < 2; ++t) {
      int r = wm + t * 16 + lhalf;
      int c = wn + t * 16 + lhalf;
      #pragma unroll
      for (int i = 0; i < 8; ++i) {
        af[t][i]     = sA[cur][r][kb0 + i];
        af[t][8 + i] = sA[cur][r][kb0 + 16 + i];
        bf[t][i]     = sB[cur][c][kb0 + i];
        bf[t][8 + i] = sB[cur][c][kb0 + 16 + i];
      }
    }
    #pragma unroll
    for (int mt = 0; mt < 2; ++mt)
      #pragma unroll
      for (int nt = 0; nt < 2; ++nt)
        acc[mt][nt] = wmma_f16(af[mt], bf[nt], acc[mt][nt]);

    if (hasNext) storeAB(cur ^ 1);
    __syncthreads();
  }

  // epilogue: add bias, store fp32
  #pragma unroll
  for (int mt = 0; mt < 2; ++mt) {
    #pragma unroll
    for (int nt = 0; nt < 2; ++nt) {
      int gn = bn + wn + nt * 16 + lhalf;
      float bv = bias[gn];
      #pragma unroll
      for (int r = 0; r < 8; ++r) {
        int gm = bm + wm + mt * 16 + ((lane >> 4) << 3) + r;
        C[(size_t)gm * N + gn] = acc[mt][nt][r] + bv;
      }
    }
  }
}

// ===========================================================================
// Kernel 2: per-head normalize + random-feature map
//   qn = scale * q / (||q|| + eps);  qp = qn @ P^T;  F = [sin(qp), cos(qp)]/8
// grid = (B*H) * (N/128);  128 rows per block
// ===========================================================================
__global__ __launch_bounds__(256)
void rfa_features(const float* __restrict__ QK, const float* __restrict__ P,
                  _Float16* __restrict__ F) {
  __shared__ _Float16 sQ[128][72];   // normalized rows (f16)
  __shared__ _Float16 sP[64][72];    // P as-is: sP[e][d]

  const int tid  = threadIdx.x;
  const int lane = tid & 31;
  const int wave = tid >> 5;
  const int nb   = blockIdx.x & 31;          // N/128 = 32 tiles
  const int bh   = blockIdx.x >> 5;
  const int b    = bh >> 4;
  const int h    = bh & 15;
  const int n0   = nb << 7;

  // load P (64x64): 16 elements per thread
  #pragma unroll
  for (int i = 0; i < 16; ++i) {
    int idx = i * 256 + tid;                 // = e*64 + d
    sP[idx >> 6][idx & 63] = (_Float16)P[idx];
  }

  // load + normalize 128 rows: 2 threads per row, 32 elems each
  {
    int r  = tid >> 1;
    int hh = (tid & 1) << 5;
    const float* src =
        QK + ((size_t)(b * N_SEQ + n0 + r)) * E_DIM + h * D_DIM + hh;
    float vals[32];
    float s = 0.f;
    #pragma unroll
    for (int i = 0; i < 32; ++i) { vals[i] = src[i]; s += vals[i] * vals[i]; }
    s += __shfl_xor(s, 1, 32);               // partner lane holds other half
    float f = 2.0f / (sqrtf(s) + EPS);       // H^0.25 = 2 folded in
    #pragma unroll
    for (int i = 0; i < 32; ++i) sQ[r][hh + i] = (_Float16)(vals[i] * f);
  }
  __syncthreads();

  // WMMA: wave handles 16 rows x 64 proj cols; K = 64 -> 2 steps
  const int lhalf = lane & 15;
  v8f acc[4] = {};
  #pragma unroll
  for (int ks = 0; ks < 2; ++ks) {
    int kb = ks * 32 + ((lane >> 4) << 3);
    v16h a;
    int rr = wave * 16 + lhalf;
    #pragma unroll
    for (int i = 0; i < 8; ++i) {
      a[i]     = sQ[rr][kb + i];
      a[8 + i] = sQ[rr][kb + 16 + i];
    }
    #pragma unroll
    for (int t = 0; t < 4; ++t) {
      v16h bf;
      int e = t * 16 + lhalf;                // B[d][e] = P[e][d] -> row e of sP
      #pragma unroll
      for (int i = 0; i < 8; ++i) {
        bf[i]     = sP[e][kb + i];
        bf[8 + i] = sP[e][kb + 16 + i];
      }
      acc[t] = wmma_f16(a, bf, acc[t]);
    }
  }

  // epilogue: sin/cos features, f16 output (B*H, N, 128)
  const float fs = 0.125f;                   // P_DIM^-0.5
  #pragma unroll
  for (int t = 0; t < 4; ++t) {
    int e = t * 16 + lhalf;
    #pragma unroll
    for (int r = 0; r < 8; ++r) {
      int m = wave * 16 + ((lane >> 4) << 3) + r;
      size_t base = ((size_t)bh * N_SEQ + n0 + m) * F_DIM;
      float qp = acc[t][r];
      F[base + e]         = (_Float16)(__sinf(qp) * fs);
      F[base + P_DIM + e] = (_Float16)(__cosf(qp) * fs);
    }
  }
}

// ===========================================================================
// Kernel 3: kv[bh] = kf[bh]^T @ v[bh]   (128 x 64, reduce over N=4096)
// one block per (b,h); 8 waves.  kf chunks staged with
// global_load_async_to_lds_b128 (ASYNCcnt), double-buffered so the async
// copy of chunk i+1 runs under the WMMAs of chunk i.
// ===========================================================================
__global__ __launch_bounds__(256)
void rfa_kv(const _Float16* __restrict__ KF, const float* __restrict__ V,
            _Float16* __restrict__ KV) {
  __shared__ _Float16 sK[2][32][136];   // kf chunk [n][e], rows 272B
  __shared__ _Float16 sV[2][32][72];    // v  chunk [n][d]

  const int tid  = threadIdx.x;
  const int lane = tid & 31;
  const int wave = tid >> 5;
  const int bh   = blockIdx.x;
  const int b    = bh >> 4;
  const int h    = bh & 15;
  const int lhalf = lane & 15;
  const int kb0   = (lane >> 4) << 3;

  v8f acc[4] = {};
  float4 rv[2];

  // async copy of kf chunk (32x128 f16 = 512 x b128, 2 per thread)
  auto kickK = [&](int buf, int nc) {
    #pragma unroll
    for (int i = 0; i < 2; ++i) {
      int idx = i * 256 + tid;
      int r   = idx >> 4;
      int cc  = (idx & 15) << 3;
      unsigned lds = (unsigned)(size_t)&sK[buf][r][cc];
      const _Float16* gp = &KF[((size_t)bh * N_SEQ + nc + r) * F_DIM + cc];
      asm volatile("global_load_async_to_lds_b128 %0, %1, off"
                   :: "v"(lds), "v"(gp) : "memory");
    }
  };
  auto loadV = [&](int nc) {
    #pragma unroll
    for (int i = 0; i < 2; ++i) {
      int idx = i * 256 + tid;
      int r   = idx >> 4;
      int c4  = (idx & 15) << 2;
      rv[i] = *(const float4*)
          &V[((size_t)(b * N_SEQ + nc + r)) * E_DIM + h * D_DIM + c4];
    }
  };
  auto storeV = [&](int buf) {
    #pragma unroll
    for (int i = 0; i < 2; ++i) {
      int idx = i * 256 + tid;
      int r   = idx >> 4;
      int c4  = (idx & 15) << 2;
      sV[buf][r][c4 + 0] = (_Float16)rv[i].x;
      sV[buf][r][c4 + 1] = (_Float16)rv[i].y;
      sV[buf][r][c4 + 2] = (_Float16)rv[i].z;
      sV[buf][r][c4 + 3] = (_Float16)rv[i].w;
    }
  };

  // prologue
  kickK(0, 0);
  loadV(0);
  storeV(0);
  asm volatile("s_wait_asynccnt 0x0" ::: "memory");
  __syncthreads();

  const int nIter = N_SEQ / 32;                  // 128
  for (int it = 0; it < nIter; ++it) {
    const int cur = it & 1;
    const bool hasNext = (it + 1) < nIter;
    if (hasNext) {
      kickK(cur ^ 1, (it + 1) * 32);             // async copy under compute
      loadV((it + 1) * 32);
    }

    // A[e][n] = kf[n][e]  (transposed read from LDS)
    v16h a;
    int e = wave * 16 + lhalf;
    #pragma unroll
    for (int i = 0; i < 8; ++i) {
      a[i]     = sK[cur][kb0 + i][e];
      a[8 + i] = sK[cur][kb0 + 16 + i][e];
    }
    #pragma unroll
    for (int t = 0; t < 4; ++t) {
      v16h bf;
      int d = t * 16 + lhalf;
      #pragma unroll
      for (int i = 0; i < 8; ++i) {
        bf[i]     = sV[cur][kb0 + i][d];
        bf[8 + i] = sV[cur][kb0 + 16 + i][d];
      }
      acc[t] = wmma_f16(a, bf, acc[t]);
    }

    if (hasNext) {
      storeV(cur ^ 1);
      asm volatile("s_wait_asynccnt 0x0" ::: "memory");
    }
    __syncthreads();
  }

  // store kv as f16: KV[bh][e][d]
  #pragma unroll
  for (int t = 0; t < 4; ++t) {
    int d = t * 16 + lhalf;
    #pragma unroll
    for (int r = 0; r < 8; ++r) {
      int e = wave * 16 + ((lane >> 4) << 3) + r;
      KV[(size_t)bh * (F_DIM * D_DIM) + e * D_DIM + d] = (_Float16)acc[t][r];
    }
  }
}

// ===========================================================================
// Kernel 4: o = qf @ kv  per (b,h); write fp32 into (B, N, E) head layout
// grid = (B*H) * (N/128); kv staged transposed in LDS once per block
// ===========================================================================
__global__ __launch_bounds__(256)
void rfa_out(const _Float16* __restrict__ QF, const _Float16* __restrict__ KV,
             float* __restrict__ O) {
  __shared__ _Float16 sKVt[64][136];  // [d][e]

  const int tid  = threadIdx.x;
  const int lane = tid & 31;
  const int wave = tid >> 5;
  const int nb   = blockIdx.x & 31;
  const int bh   = blockIdx.x >> 5;
  const int b    = bh >> 4;
  const int h    = bh & 15;
  const int n0   = nb << 7;
  const int lhalf = lane & 15;

  // stage kv transposed: 8192 halves, 32 per thread
  #pragma unroll
  for (int i = 0; i < 32; ++i) {
    int idx = i * 256 + tid;              // = e*64 + d
    sKVt[idx & 63][idx >> 6] = KV[(size_t)bh * (F_DIM * D_DIM) + idx];
  }
  __syncthreads();

  v8f acc[4] = {};
  const _Float16* qrows = QF + ((size_t)bh * N_SEQ + n0) * F_DIM;

  #pragma unroll
  for (int ks = 0; ks < 4; ++ks) {
    int kb = ks * 32 + ((lane >> 4) << 3);
    // A fragment straight from global (qf rows are contiguous in k)
    int m = wave * 16 + lhalf;
    const _Float16* p = qrows + (size_t)m * F_DIM;
    half8 lo = *(const half8*)&p[kb];
    half8 hi = *(const half8*)&p[kb + 16];
    v16h a;
    #pragma unroll
    for (int i = 0; i < 8; ++i) { a[i] = lo[i]; a[8 + i] = hi[i]; }
    #pragma unroll
    for (int t = 0; t < 4; ++t) {
      v16h bf;
      int d = t * 16 + lhalf;
      #pragma unroll
      for (int i = 0; i < 8; ++i) {
        bf[i]     = sKVt[d][kb + i];
        bf[8 + i] = sKVt[d][kb + 16 + i];
      }
      acc[t] = wmma_f16(a, bf, acc[t]);
    }
  }

  // scatter into (B, N, E)
  #pragma unroll
  for (int t = 0; t < 4; ++t) {
    int d = t * 16 + lhalf;
    #pragma unroll
    for (int r = 0; r < 8; ++r) {
      int m = wave * 16 + ((lane >> 4) << 3) + r;
      O[((size_t)(b * N_SEQ + n0 + m)) * E_DIM + h * D_DIM + d] = acc[t][r];
    }
  }
}

// ===========================================================================
// launch
// ===========================================================================
extern "C" void kernel_launch(void* const* d_in, const int* in_sizes, int n_in,
                              void* d_out, int out_size, void* d_ws, size_t ws_size,
                              hipStream_t stream) {
  (void)in_sizes; (void)n_in; (void)out_size; (void)ws_size;

  const float* x  = (const float*)d_in[0];
  const float* Wq = (const float*)d_in[1];
  const float* bq = (const float*)d_in[2];
  const float* Wk = (const float*)d_in[3];
  const float* bk = (const float*)d_in[4];
  const float* Wv = (const float*)d_in[5];
  const float* bv = (const float*)d_in[6];
  const float* Wo = (const float*)d_in[7];
  const float* bo = (const float*)d_in[8];
  const float* P  = (const float*)d_in[9];
  float* out = (float*)d_out;

  const int M = B_SZ * N_SEQ;                 // 16384
  const size_t MN = (size_t)M * E_DIM;        // 16.7M floats
  const size_t FEAT = (size_t)B_SZ * H_NUM * N_SEQ * F_DIM;  // 33.5M halves

  float*     q  = (float*)d_ws;
  float*     k  = q + MN;
  float*     v  = k + MN;
  _Float16*  qf = (_Float16*)(v + MN);
  _Float16*  kf = qf + FEAT;
  _Float16*  kv = kf + FEAT;
  float*     o  = q;                          // reuse q buffer for o

  const dim3 blk(256);
  const int gemmGrid = (M / 128) * (E_DIM / 64);     // 2048
  const int featGrid = B_SZ * H_NUM * (N_SEQ / 128); // 2048

  // 1. Q/K/V projections (WMMA GEMMs)
  gemm_xWt_bias<<<gemmGrid, blk, 0, stream>>>(x, Wq, bq, q, M, E_DIM, E_DIM);
  gemm_xWt_bias<<<gemmGrid, blk, 0, stream>>>(x, Wk, bk, k, M, E_DIM, E_DIM);
  gemm_xWt_bias<<<gemmGrid, blk, 0, stream>>>(x, Wv, bv, v, M, E_DIM, E_DIM);

  // 2. random-feature maps
  rfa_features<<<featGrid, blk, 0, stream>>>(q, P, qf);
  rfa_features<<<featGrid, blk, 0, stream>>>(k, P, kf);

  // 3. kv = kf^T @ v per head (async-to-LDS staging)
  rfa_kv<<<B_SZ * H_NUM, blk, 0, stream>>>(kf, v, kv);

  // 4. o = qf @ kv, scattered back to (B, N, E)
  rfa_out<<<featGrid, blk, 0, stream>>>(qf, kv, o);

  // 5. output projection
  gemm_xWt_bias<<<gemmGrid, blk, 0, stream>>>(o, Wo, bo, out, M, E_DIM, E_DIM);
}
